// RNNDecoder_46952582480486
// MI455X (gfx1250) — matrix operand order repaired and checked by reading, not compiled, not run on hardware
//
#include <hip/hip_runtime.h>
#include <math.h>

// ---------------------------------------------------------------------------
// Autoregressive GRU decoder for MI455X (gfx1250, wave32, WMMA).
// Strategy: convert weights fp32->bf16 once (L2-resident: ~75 MB < 192 MB L2),
// run all GEMMs with V_WMMA_F32_16X16X32_BF16 (f32 accumulate), per-step
// kernels chained on the stream (serial token feedback).
// ---------------------------------------------------------------------------

typedef __bf16 bf16_t;
typedef __attribute__((ext_vector_type(16))) __bf16 v16bf;
typedef __attribute__((ext_vector_type(8)))  float  v8f;

#define BATCH   32
#define EMBED   512
#define HIDDEN  1024
#define G3H     (3 * HIDDEN)
#define VOCAB   32000
#define TSTEPS  64
#define SOS_TOK 1

// ---- WMMA fragment helpers (wave32 layouts per CDNA5 ISA 7.12.2) ----------

__device__ __forceinline__ v16bf load2x16(const bf16_t* p0, const bf16_t* p1) {
  union { v16bf v; uint4 q[2]; } u;
  u.q[0] = *(const uint4*)p0;
  u.q[1] = *(const uint4*)p1;
  return u.v;
}

// A fragment: 16x32 bf16 tile of row-major [M x K]; rows mbase..mbase+15.
// lane<16 : elems 0..7 = A[lr][k0+0..7],  8..15 = A[lr][k0+16..23]
// lane>=16: elems 0..7 = A[lr][k0+8..15], 8..15 = A[lr][k0+24..31]
__device__ __forceinline__ v16bf loadA(const bf16_t* a, int K, int mbase,
                                       int k0, int lr, int half) {
  const bf16_t* p = a + (size_t)(mbase + lr) * K + k0 + half * 8;
  return load2x16(p, p + 16);
}

// B fragment: 32x16 tile with B[k][n] = W[n][k], W row-major [N x K].
// lane<16 : K=k0+0..7,16..23 of column (ncol0+lr); lane>=16: K=k0+8..15,24..31.
__device__ __forceinline__ v16bf loadB(const bf16_t* w, int K, int ncol0,
                                       int k0, int lr, int half) {
  const bf16_t* p = w + (size_t)(ncol0 + lr) * K + k0 + half * 8;
  return load2x16(p, p + 16);
}

__device__ __forceinline__ v8f wmma_bf16(v16bf a, v16bf b, v8f c) {
  // (neg_a, A, neg_b, B, c_mod, C, reuse_a, reuse_b)
  return __builtin_amdgcn_wmma_f32_16x16x32_bf16(false, a, false, b,
                                                 (short)0, c, false, false);
}

// ---- setup kernels --------------------------------------------------------

__global__ void k_f32_to_bf16(const float* __restrict__ in,
                              bf16_t* __restrict__ out, int n) {
  int i = blockIdx.x * blockDim.x + threadIdx.x;
  int stride = gridDim.x * blockDim.x;
  for (; i < n; i += stride) out[i] = (bf16_t)in[i];
}

__global__ void k_init_state(const float* __restrict__ h0,
                             float* __restrict__ hF, bf16_t* __restrict__ hB,
                             int* __restrict__ tok) {
  int i = blockIdx.x * blockDim.x + threadIdx.x;
  if (i < BATCH * HIDDEN) {
    float v = h0[i];
    hF[i] = v;
    hB[i] = (bf16_t)v;
  }
  if (i < BATCH) tok[i] = SOS_TOK;
}

// ---- per-step kernels -----------------------------------------------------

// x = tanh(embedding[tok])  -> bf16, [32 x 512]
__global__ void k_embed_tanh(const float* __restrict__ emb,
                             const int* __restrict__ tok,
                             bf16_t* __restrict__ x) {
  int b = blockIdx.x;           // 32 blocks
  int e = threadIdx.x;          // 512 threads
  float v = emb[(size_t)tok[b] * EMBED + e];
  x[b * EMBED + e] = (bf16_t)tanhf(v);
}

// gi = x @ w_ih^T + b_ih ; gh = h @ w_hh^T + b_hh    (both [32 x 3072], f32)
// grid: 24 blocks x 256 threads = 192 waves, one N-tile (16 cols) per wave.
__global__ void k_gates(const bf16_t* __restrict__ x, const bf16_t* __restrict__ h,
                        const bf16_t* __restrict__ wih, const bf16_t* __restrict__ whh,
                        const float* __restrict__ bih, const float* __restrict__ bhh,
                        float* __restrict__ gi, float* __restrict__ gh) {
  int wave = threadIdx.x >> 5;
  int lane = threadIdx.x & 31;
  int lr = lane & 15, half = lane >> 4;
  int ntile = blockIdx.x * 8 + wave;          // 0..191
  int col = ntile * 16 + lr;

  v8f ai0 = {}, ai1 = {}, ah0 = {}, ah1 = {};
  for (int k0 = 0; k0 < EMBED; k0 += 32) {
    v16bf b  = loadB(wih, EMBED, ntile * 16, k0, lr, half);
    v16bf a0 = loadA(x, EMBED, 0,  k0, lr, half);
    v16bf a1 = loadA(x, EMBED, 16, k0, lr, half);
    ai0 = wmma_bf16(a0, b, ai0);
    ai1 = wmma_bf16(a1, b, ai1);
  }
  for (int k0 = 0; k0 < HIDDEN; k0 += 32) {
    v16bf b  = loadB(whh, HIDDEN, ntile * 16, k0, lr, half);
    v16bf a0 = loadA(h, HIDDEN, 0,  k0, lr, half);
    v16bf a1 = loadA(h, HIDDEN, 16, k0, lr, half);
    ah0 = wmma_bf16(a0, b, ah0);
    ah1 = wmma_bf16(a1, b, ah1);
  }
  float vbi = bih[col], vbh = bhh[col];
#pragma unroll
  for (int j = 0; j < 8; ++j) {
    int r0 = half * 8 + j;          // C/D layout: VGPR j -> M = j + 8*half
    int r1 = 16 + r0;
    gi[r0 * G3H + col] = ai0[j] + vbi;
    gi[r1 * G3H + col] = ai1[j] + vbi;
    gh[r0 * G3H + col] = ah0[j] + vbh;
    gh[r1 * G3H + col] = ah1[j] + vbh;
  }
}

// GRU elementwise update.  32 blocks x 1024 threads.
__global__ void k_gru(const float* __restrict__ gi, const float* __restrict__ gh,
                      float* __restrict__ hF, bf16_t* __restrict__ hB) {
  int b = blockIdx.x, j = threadIdx.x;
  const float* gib = gi + b * G3H;
  const float* ghb = gh + b * G3H;
  float ir = gib[j], iz = gib[HIDDEN + j], in_ = gib[2 * HIDDEN + j];
  float hr = ghb[j], hz = ghb[HIDDEN + j], hn  = ghb[2 * HIDDEN + j];
  float r = 1.f / (1.f + expf(-(ir + hr)));
  float z = 1.f / (1.f + expf(-(iz + hz)));
  float n = tanhf(in_ + r * hn);
  float hold = hF[b * HIDDEN + j];
  float hnew = (1.f - z) * n + z * hold;
  hF[b * HIDDEN + j] = hnew;
  hB[b * HIDDEN + j] = (bf16_t)hnew;
}

// logits = h @ w_out^T + b_out   [32 x 32000]
// grid: 250 blocks x 256 threads = 2000 waves, one 16-wide V-tile per wave.
__global__ void k_logits(const bf16_t* __restrict__ h, const bf16_t* __restrict__ wout,
                         const float* __restrict__ bout, float* __restrict__ logits) {
  int wave = threadIdx.x >> 5;
  int lane = threadIdx.x & 31;
  int lr = lane & 15, half = lane >> 4;
  int vtile = blockIdx.x * 8 + wave;          // 0..1999

  v8f c0 = {}, c1 = {};
  for (int k0 = 0; k0 < HIDDEN; k0 += 32) {
    v16bf b  = loadB(wout, HIDDEN, vtile * 16, k0, lr, half);
    v16bf a0 = loadA(h, HIDDEN, 0,  k0, lr, half);
    v16bf a1 = loadA(h, HIDDEN, 16, k0, lr, half);
    c0 = wmma_bf16(a0, b, c0);
    c1 = wmma_bf16(a1, b, c1);
  }
  int col = vtile * 16 + lr;
  float bb = bout[col];
#pragma unroll
  for (int j = 0; j < 8; ++j) {
    int r0 = half * 8 + j;
    logits[(size_t)r0 * VOCAB + col]        = c0[j] + bb;
    logits[(size_t)(16 + r0) * VOCAB + col] = c1[j] + bb;
  }
}

// log_softmax + argmax (first-max tie-break) per batch row; writes d_out slice
// and the next-token buffer.  32 blocks x 256 threads.
__global__ void k_logsoftmax(const float* __restrict__ logits,
                             float* __restrict__ out, int* __restrict__ tok) {
  __shared__ float sm[256];
  __shared__ int   si[256];
  int b = blockIdx.x, tid = threadIdx.x;
  const float* row = logits + (size_t)b * VOCAB;

  float m = -3.4e38f; int mi = 0;
  for (int v = tid; v < VOCAB; v += 256) {
    float l = row[v];
    if (l > m) { m = l; mi = v; }     // strict '>' keeps first occurrence
  }
  sm[tid] = m; si[tid] = mi;
  __syncthreads();
  for (int s = 128; s > 0; s >>= 1) {
    if (tid < s) {
      float m2 = sm[tid + s]; int i2 = si[tid + s];
      if (m2 > sm[tid] || (m2 == sm[tid] && i2 < si[tid])) { sm[tid] = m2; si[tid] = i2; }
    }
    __syncthreads();
  }
  float rowmax = sm[0];
  int   rowarg = si[0];
  __syncthreads();

  float s = 0.f;
  for (int v = tid; v < VOCAB; v += 256) s += expf(row[v] - rowmax);
  sm[tid] = s;
  __syncthreads();
  for (int st = 128; st > 0; st >>= 1) {
    if (tid < st) sm[tid] += sm[tid + st];
    __syncthreads();
  }
  float logZ = rowmax + logf(sm[0]);

  for (int v = tid; v < VOCAB; v += 256)
    out[(size_t)b * VOCAB + v] = row[v] - logZ;
  if (tid == 0) tok[b] = rowarg;
}

// ---------------------------------------------------------------------------

extern "C" void kernel_launch(void* const* d_in, const int* in_sizes, int n_in,
                              void* d_out, int out_size, void* d_ws, size_t ws_size,
                              hipStream_t stream) {
  (void)in_sizes; (void)n_in; (void)out_size; (void)ws_size;

  const float* init_hidden = (const float*)d_in[0];
  // d_in[1] encoder_outputs: unused by the reference (no attention applied)
  // d_in[2] tgt_len: traced constant 64
  const float* embedding = (const float*)d_in[3];
  const float* w_ih  = (const float*)d_in[4];
  const float* b_ih  = (const float*)d_in[5];
  const float* w_hh  = (const float*)d_in[6];
  const float* b_hh  = (const float*)d_in[7];
  const float* w_out = (const float*)d_in[8];
  const float* b_out = (const float*)d_in[9];
  float* out = (float*)d_out;

  // workspace carve-up (256B aligned), total ~80.1 MB
  char* ws = (char*)d_ws;
  size_t off = 0;
  auto take = [&](size_t bytes) -> char* {
    char* p = ws + off;
    off = (off + bytes + 255) & ~(size_t)255;
    return p;
  };
  bf16_t* wih_b  = (bf16_t*)take((size_t)G3H * EMBED * sizeof(bf16_t));
  bf16_t* whh_b  = (bf16_t*)take((size_t)G3H * HIDDEN * sizeof(bf16_t));
  bf16_t* wout_b = (bf16_t*)take((size_t)VOCAB * HIDDEN * sizeof(bf16_t));
  float*  hF     = (float*) take((size_t)BATCH * HIDDEN * sizeof(float));
  bf16_t* hB     = (bf16_t*)take((size_t)BATCH * HIDDEN * sizeof(bf16_t));
  bf16_t* xB     = (bf16_t*)take((size_t)BATCH * EMBED * sizeof(bf16_t));
  float*  gi     = (float*) take((size_t)BATCH * G3H * sizeof(float));
  float*  gh     = (float*) take((size_t)BATCH * G3H * sizeof(float));
  float*  logits = (float*) take((size_t)BATCH * VOCAB * sizeof(float));
  int*    tok    = (int*)   take((size_t)BATCH * sizeof(int));

  // one-time (per launch) weight conversion + state init
  k_f32_to_bf16<<<512, 256, 0, stream>>>(w_ih, wih_b, G3H * EMBED);
  k_f32_to_bf16<<<512, 256, 0, stream>>>(w_hh, whh_b, G3H * HIDDEN);
  k_f32_to_bf16<<<4096, 256, 0, stream>>>(w_out, wout_b, VOCAB * HIDDEN);
  k_init_state<<<(BATCH * HIDDEN + 255) / 256, 256, 0, stream>>>(init_hidden, hF, hB, tok);

  // serial greedy decode
  for (int t = 0; t < TSTEPS; ++t) {
    k_embed_tanh<<<BATCH, EMBED, 0, stream>>>(embedding, tok, xB);
    k_gates<<<24, 256, 0, stream>>>(xB, hB, wih_b, whh_b, b_ih, b_hh, gi, gh);
    k_gru<<<BATCH, HIDDEN, 0, stream>>>(gi, gh, hF, hB);
    k_logits<<<250, 256, 0, stream>>>(hB, wout_b, b_out, logits);
    k_logsoftmax<<<BATCH, 256, 0, stream>>>(
        logits, out + (size_t)t * BATCH * VOCAB, tok);
  }
}
